// TokenRoutedMLP_85598698209845
// MI455X (gfx1250) — matrix operand order repaired and checked by reading, not compiled, not run on hardware
//
#include <hip/hip_runtime.h>
#include <hip/hip_bf16.h>

typedef __bf16 bf16_t;
typedef __attribute__((ext_vector_type(16))) __bf16 v16bf;
typedef __attribute__((ext_vector_type(8)))  float  v8f;
typedef __attribute__((ext_vector_type(4)))  int    v4i;

#define DIMD     2048
#define TWO_I    2048
#define INTER    1024
#define NEXP     8
#define T_TOK    8192
#define BM       128
#define BK       32
#define MAX_TILES 64          // ceil(8192/128)
#define LDS_STRIDE 34         // 17 banks/row, coprime with 64 -> conflict-free frag reads
#define A2_STRIDE  40         // gemm2 A: 80B rows -> 16B-aligned b128 async stores

// workspace ints: [0..7] counts, [8..15] cursors, [16..24] compact offsets,
//                 [25..33] padded offsets, [64..8255] sorted token ids
// byte offset 65536: h buffer, bf16 [<=9216][1024] (expert segments padded to 128 rows)
#define WS_SORT  64
#define WS_H_OFF 65536

#if defined(__has_builtin)
#if __has_builtin(__builtin_amdgcn_global_load_async_to_lds_b128) && \
    __has_builtin(__builtin_amdgcn_s_wait_asynccnt)
#define HAVE_ASYNC_LDS 1
#endif
#endif
#ifndef HAVE_ASYNC_LDS
#define HAVE_ASYNC_LDS 0
#endif

#if HAVE_ASYNC_LDS
typedef __attribute__((address_space(1))) v4i* as1_v4i;
typedef __attribute__((address_space(3))) v4i* as3_v4i;
#endif

union FragU { v16bf v; unsigned int u[8]; };

__device__ __forceinline__ unsigned int pk_bf16(float a, float b) {
    union { __bf16 h[2]; unsigned int u; } r;
    r.h[0] = (__bf16)a; r.h[1] = (__bf16)b;
    return r.u;
}

// ---------------- routing ----------------
__global__ void moe_route_init(int* wsi) {
    if (threadIdx.x < 16) wsi[threadIdx.x] = 0;
}
__global__ void moe_route_count(const int* __restrict__ ids, int* __restrict__ wsi) {
    int t = blockIdx.x * blockDim.x + threadIdx.x;
    if (t < T_TOK) atomicAdd(&wsi[ids[t]], 1);
}
__global__ void moe_route_offsets(int* wsi) {
    if (threadIdx.x == 0) {
        int acc = 0, pacc = 0;
        for (int e = 0; e < NEXP; ++e) {
            wsi[16 + e] = acc;      // compact offset (sorted-id list)
            wsi[8 + e]  = acc;      // cursor
            wsi[25 + e] = pacc;     // padded offset (h rows)
            acc  += wsi[e];
            pacc += ((wsi[e] + BM - 1) / BM) * BM;
        }
        wsi[16 + NEXP] = acc;
        wsi[25 + NEXP] = pacc;
    }
}
__global__ void moe_route_scatter(const int* __restrict__ ids, int* __restrict__ wsi) {
    int t = blockIdx.x * blockDim.x + threadIdx.x;
    if (t < T_TOK) {
        int e   = ids[t];
        int pos = atomicAdd(&wsi[8 + e], 1);
        wsi[WS_SORT + pos] = t;
    }
}

// ---------------- GEMM1: gu = x @ W1[e]; h = silu(g)*u (bf16, padded segments) ----------------
__global__ __launch_bounds__(256) void moe_gemm1(
    const float* __restrict__ x, const float* __restrict__ gup,
    const int* __restrict__ wsi, bf16_t* __restrict__ hbuf) {

    __shared__ bf16_t Als[BM][LDS_STRIDE];       // [row][K]
    __shared__ bf16_t Bls[128][LDS_STRIDE];      // [col][K]: cols 0..63 = g, 64..127 = u
    __shared__ int rowTok[BM];

    const int bx = blockIdx.x;                   // h-col block: cols bx*64..+63
    const int e  = blockIdx.y >> 6;
    const int mt = blockIdx.y & (MAX_TILES - 1);
    const int cnt = wsi[e];
    if (mt * BM >= cnt) return;
    const int segBase = wsi[16 + e];
    const int pseg    = wsi[25 + e];

    const int tid = threadIdx.x;
    if (tid < BM) {
        int r = mt * BM + tid;
        rowTok[tid] = (r < cnt) ? wsi[WS_SORT + segBase + r] : -1;
    }
    __syncthreads();

    const int lane = tid & 31, wid = tid >> 5;
    const int waveM = wid & 3, waveN = wid >> 2;
    const int kh = lane >> 4, l16 = lane & 15;

    v8f accg[2][2], accu[2][2];
    for (int i = 0; i < 2; ++i)
        for (int j = 0; j < 2; ++j)
            for (int r = 0; r < 8; ++r) { accg[i][j][r] = 0.f; accu[i][j][r] = 0.f; }

    const float* wbase = gup + (size_t)e * DIMD * TWO_I;
    const int cg = bx * 64, cu = INTER + bx * 64;

    for (int k0 = 0; k0 < DIMD; k0 += BK) {
        // prefetch next K-tile of W1 (one cacheline per thread)
        if (k0 + BK < DIMD) {
            int kp = tid >> 5, c = (tid & 31) * 4;
            int scol = (c < 64) ? (cg + c) : (cu + c - 64);
            __builtin_prefetch((const void*)(wbase + (size_t)(k0 + BK + 2 * kp) * TWO_I + scol), 0, 0);
        }
        // A tile: 128x32 fp32 -> packed bf16 (gathered rows). 1024 float4, 4/thread.
        #pragma unroll
        for (int i = 0; i < 4; ++i) {
            int idx = i * 256 + tid;
            int r = idx >> 3, c = (idx & 7) * 4;
            int tok = rowTok[r];
            float4 v = make_float4(0.f, 0.f, 0.f, 0.f);
            if (tok >= 0) v = *(const float4*)(x + (size_t)tok * DIMD + k0 + c);
            *(unsigned int*)&Als[r][c]     = pk_bf16(v.x, v.y);
            *(unsigned int*)&Als[r][c + 2] = pk_bf16(v.z, v.w);
        }
        // B tile: K-row pairs so bf16 packs along K; stored [col][K]. 512 items, 2/thread.
        #pragma unroll
        for (int i = 0; i < 2; ++i) {
            int idx = i * 256 + tid;
            int kp = idx >> 5;            // k pair 0..15
            int c  = (idx & 31) * 4;      // col group
            int scol = (c < 64) ? (cg + c) : (cu + c - 64);
            const float* s = wbase + (size_t)(k0 + 2 * kp) * TWO_I + scol;
            float4 v0 = *(const float4*)s;
            float4 v1 = *(const float4*)(s + TWO_I);
            *(unsigned int*)&Bls[c + 0][2 * kp] = pk_bf16(v0.x, v1.x);
            *(unsigned int*)&Bls[c + 1][2 * kp] = pk_bf16(v0.y, v1.y);
            *(unsigned int*)&Bls[c + 2][2 * kp] = pk_bf16(v0.z, v1.z);
            *(unsigned int*)&Bls[c + 3][2 * kp] = pk_bf16(v0.w, v1.w);
        }
        __syncthreads();

        v16bf a[2];
        #pragma unroll
        for (int ms = 0; ms < 2; ++ms) {
            int m = waveM * 32 + ms * 16 + l16;
            FragU f;
            const unsigned int* p0 = (const unsigned int*)&Als[m][kh * 16];
            const unsigned int* p1 = (const unsigned int*)&Als[m][kh * 16 + 8];
            #pragma unroll
            for (int j = 0; j < 4; ++j) { f.u[j] = p0[j]; f.u[4 + j] = p1[j]; }
            a[ms] = f.v;
        }
        v16bf bg[2], bu[2];
        #pragma unroll
        for (int ns = 0; ns < 2; ++ns) {
            int n = waveN * 32 + ns * 16 + l16;
            FragU fg, fu;
            const unsigned int* pg = (const unsigned int*)&Bls[n][kh * 16];
            const unsigned int* pu = (const unsigned int*)&Bls[64 + n][kh * 16];
            #pragma unroll
            for (int j = 0; j < 8; ++j) { fg.u[j] = pg[j]; fu.u[j] = pu[j]; }
            bg[ns] = fg.v; bu[ns] = fu.v;
        }
        #pragma unroll
        for (int ms = 0; ms < 2; ++ms)
            #pragma unroll
            for (int ns = 0; ns < 2; ++ns) {
                accg[ms][ns] = __builtin_amdgcn_wmma_f32_16x16x32_bf16(
                    false, a[ms], false, bg[ns], (short)0, accg[ms][ns], false, false);
                accu[ms][ns] = __builtin_amdgcn_wmma_f32_16x16x32_bf16(
                    false, a[ms], false, bu[ns], (short)0, accu[ms][ns], false, false);
            }
        __syncthreads();
    }

    // h = silu(g)*u; pad rows compute exact zeros -> store unconditionally (zero-fills padding)
    #pragma unroll
    for (int ms = 0; ms < 2; ++ms)
        #pragma unroll
        for (int ns = 0; ns < 2; ++ns)
            #pragma unroll
            for (int r = 0; r < 8; ++r) {
                int rowRel = waveM * 32 + ms * 16 + r + 8 * kh;
                float g = accg[ms][ns][r], u = accu[ms][ns][r];
                float hv = (g / (1.f + __expf(-g))) * u;
                int p   = pseg + mt * BM + rowRel;
                int col = bx * 64 + waveN * 32 + ns * 16 + l16;
                hbuf[(size_t)p * INTER + col] = (bf16_t)hv;
            }
}

// ---------------- GEMM2: y = h @ W2[e], scatter fp32 to out[token] ----------------
__global__ __launch_bounds__(256) void moe_gemm2(
    const bf16_t* __restrict__ hbuf, const float* __restrict__ wdown,
    const int* __restrict__ wsi, float* __restrict__ out) {

    __shared__ bf16_t Als[BM][A2_STRIDE];        // 80B rows: 16B-aligned async b128 dests
    __shared__ bf16_t Bls[128][LDS_STRIDE];
    __shared__ int rowTok[BM];

    const int bx = blockIdx.x;                   // out cols cb = bx*128
    const int e  = blockIdx.y >> 6;
    const int mt = blockIdx.y & (MAX_TILES - 1);
    const int cnt = wsi[e];
    if (mt * BM >= cnt) return;
    const int segBase = wsi[16 + e];
    const int pseg    = wsi[25 + e];

    const int tid = threadIdx.x;
    if (tid < BM) {
        int r = mt * BM + tid;
        rowTok[tid] = (r < cnt) ? wsi[WS_SORT + segBase + r] : -1;
    }
    __syncthreads();

    const int lane = tid & 31, wid = tid >> 5;
    const int waveM = wid & 3, waveN = wid >> 2;
    const int kh = lane >> 4, l16 = lane & 15;

    v8f acc[2][4];
    for (int i = 0; i < 2; ++i)
        for (int j = 0; j < 4; ++j)
            for (int r = 0; r < 8; ++r) acc[i][j][r] = 0.f;

    const float* wbase = wdown + (size_t)e * INTER * DIMD;
    const int cb = bx * 128;

    for (int k0 = 0; k0 < INTER; k0 += BK) {
        if (k0 + BK < INTER) {
            int kp = tid >> 5, c = (tid & 31) * 4;
            __builtin_prefetch((const void*)(wbase + (size_t)(k0 + BK + 2 * kp) * DIMD + cb + c), 0, 0);
        }
        // A tile: 128x32 bf16, h rows are padded+zero-filled -> unconditional async copy.
        #pragma unroll
        for (int i = 0; i < 2; ++i) {
            int idx = i * 256 + tid;
            int r = idx >> 2, c = (idx & 3) * 8;
            const bf16_t* src = hbuf + (size_t)(pseg + mt * BM + r) * INTER + k0 + c;
#if HAVE_ASYNC_LDS
            __builtin_amdgcn_global_load_async_to_lds_b128(
                (as1_v4i)src, (as3_v4i)&Als[r][c], 0, 0);
#else
            uint4 v = *(const uint4*)src;
            unsigned int* dst = (unsigned int*)&Als[r][c];
            dst[0] = v.x; dst[1] = v.y; dst[2] = v.z; dst[3] = v.w;
#endif
        }
        // B tile: 32x128 fp32 W2 -> packed bf16, stored [col][K]. K-pairs, 2/thread.
        #pragma unroll
        for (int i = 0; i < 2; ++i) {
            int idx = i * 256 + tid;
            int kp = idx >> 5;
            int c  = (idx & 31) * 4;
            const float* s = wbase + (size_t)(k0 + 2 * kp) * DIMD + cb + c;
            float4 v0 = *(const float4*)s;
            float4 v1 = *(const float4*)(s + DIMD);
            *(unsigned int*)&Bls[c + 0][2 * kp] = pk_bf16(v0.x, v1.x);
            *(unsigned int*)&Bls[c + 1][2 * kp] = pk_bf16(v0.y, v1.y);
            *(unsigned int*)&Bls[c + 2][2 * kp] = pk_bf16(v0.z, v1.z);
            *(unsigned int*)&Bls[c + 3][2 * kp] = pk_bf16(v0.w, v1.w);
        }
#if HAVE_ASYNC_LDS
        __builtin_amdgcn_s_wait_asynccnt(0);
#endif
        __syncthreads();

        v16bf a[2];
        #pragma unroll
        for (int ms = 0; ms < 2; ++ms) {
            int m = waveM * 32 + ms * 16 + l16;
            FragU f;
            const unsigned int* p0 = (const unsigned int*)&Als[m][kh * 16];
            const unsigned int* p1 = (const unsigned int*)&Als[m][kh * 16 + 8];
            #pragma unroll
            for (int j = 0; j < 4; ++j) { f.u[j] = p0[j]; f.u[4 + j] = p1[j]; }
            a[ms] = f.v;
        }
        v16bf b[4];
        #pragma unroll
        for (int ns = 0; ns < 4; ++ns) {
            int n = waveN * 64 + ns * 16 + l16;
            FragU f;
            const unsigned int* pb = (const unsigned int*)&Bls[n][kh * 16];
            #pragma unroll
            for (int j = 0; j < 8; ++j) f.u[j] = pb[j];
            b[ns] = f.v;
        }
        #pragma unroll
        for (int ms = 0; ms < 2; ++ms)
            #pragma unroll
            for (int ns = 0; ns < 4; ++ns)
                acc[ms][ns] = __builtin_amdgcn_wmma_f32_16x16x32_bf16(
                    false, a[ms], false, b[ns], (short)0, acc[ms][ns], false, false);
        __syncthreads();
    }

    #pragma unroll
    for (int ms = 0; ms < 2; ++ms)
        #pragma unroll
        for (int ns = 0; ns < 4; ++ns)
            #pragma unroll
            for (int r = 0; r < 8; ++r) {
                int rowRel = waveM * 32 + ms * 16 + r + 8 * kh;
                int tok = rowTok[rowRel];
                if (tok >= 0) {
                    int col = cb + waveN * 64 + ns * 16 + l16;
                    out[(size_t)tok * DIMD + col] = acc[ms][ns][r];
                }
            }
}

extern "C" void kernel_launch(void* const* d_in, const int* in_sizes, int n_in,
                              void* d_out, int out_size, void* d_ws, size_t ws_size,
                              hipStream_t stream) {
    const float* x    = (const float*)d_in[0];
    const int*   ids  = (const int*)d_in[1];
    const float* gup  = (const float*)d_in[2];
    const float* dwn  = (const float*)d_in[3];
    float* out        = (float*)d_out;
    int*   wsi        = (int*)d_ws;
    bf16_t* hbuf      = (bf16_t*)((char*)d_ws + WS_H_OFF);

    moe_route_init<<<1, 64, 0, stream>>>(wsi);
    moe_route_count<<<T_TOK / 256, 256, 0, stream>>>(ids, wsi);
    moe_route_offsets<<<1, 32, 0, stream>>>(wsi);
    moe_route_scatter<<<T_TOK / 256, 256, 0, stream>>>(ids, wsi);

    dim3 grid(16, NEXP * MAX_TILES);
    moe_gemm1<<<grid, 256, 0, stream>>>(x, gup, wsi, hbuf);
    moe_gemm2<<<grid, 256, 0, stream>>>(hbuf, dwn, wsi, out);
}
